// LSTM_33277406609630
// MI455X (gfx1250) — matrix operand order repaired and checked by reading, not compiled, not run on hardware
//
#include <hip/hip_runtime.h>
#include <hip/hip_bf16.h>
#include <stdint.h>

typedef __attribute__((ext_vector_type(16))) __bf16 v16bf;
typedef __attribute__((ext_vector_type(8)))  float  v8f;

#define I_SIZE 1024
#define H_SIZE 1024
#define BATCH  64
#define SEQ    512
#define MTOT   (BATCH * SEQ)   /* 32768 rows of x */
#define N4H    (4 * H_SIZE)    /* 4096 gate columns */

// ---------- helpers ----------

__device__ __forceinline__ unsigned short f2bf(float f) {
  unsigned int u = __float_as_uint(f);
  u += 0x7FFFu + ((u >> 16) & 1u);        // round-to-nearest-even
  return (unsigned short)(u >> 16);
}

__device__ __forceinline__ float sigmoid_fast(float x) {
  return 1.0f / (1.0f + __expf(-x));
}
__device__ __forceinline__ float tanh_fast(float x) {
  float e = __expf(-2.0f * x);
  return (1.0f - e) / (1.0f + e);
}

// A-matrix (16x32, bf16) per-lane element->K map (ISA 7.12.2):
// lane<16: e0..7 -> K0..7, e8..15 -> K16..23 ; lane>=16: +8
__device__ __forceinline__ int kofA(int lane, int e) {
  return e + ((e >= 8) ? 8 : 0) + ((lane >= 16) ? 8 : 0);
}

__device__ __forceinline__ v8f wmma_bf16(v16bf a, v16bf b, v8f c) {
  return __builtin_amdgcn_wmma_f32_16x16x32_bf16(
      /*neg_a=*/false, a, /*neg_b=*/false, b,
      /*c_mod=*/(short)0, c, /*reuse_a=*/false, /*reuse_b=*/false);
}

// CDNA5 async memory->LDS copy (GLOBAL_LOAD_ASYNC_TO_LDS_B128, ASYNCcnt).
// lds_off = workgroup-relative LDS byte offset; gaddr = per-lane global addr.
__device__ __forceinline__ void async_ld_lds_b128(unsigned int lds_off,
                                                  const void* gaddr) {
  asm volatile("global_load_async_to_lds_b128 %0, %1, off"
               :: "v"(lds_off), "v"((unsigned long long)(size_t)gaddr)
               : "memory");
}
__device__ __forceinline__ void wait_asynccnt0() {
  asm volatile("s_wait_asynccnt 0" ::: "memory");
}

__device__ __forceinline__ float bias_sum(int n,
    const float* b_ii, const float* b_hi, const float* b_if, const float* b_hf,
    const float* b_ig, const float* b_hg, const float* b_io, const float* b_ho) {
  int gate = n >> 10, col = n & 1023;
  const float* bi; const float* bh;
  if      (gate == 0) { bi = b_ii; bh = b_hi; }
  else if (gate == 1) { bi = b_if; bh = b_hf; }
  else if (gate == 2) { bi = b_ig; bh = b_hg; }
  else                { bi = b_io; bh = b_ho; }
  return bi[col] + bh[col];
}

// ---------- packing kernels ----------

// Pack x (f32 row-major MTOTxI) into bf16 A-tiles. tile_id = mt*32 + kt.
__global__ void pack_x_kernel(const float* __restrict__ x,
                              unsigned short* __restrict__ dst) {
  int tid  = blockIdx.x * blockDim.x + threadIdx.x;   // 2,097,152 threads
  int lane = tid & 31;
  int tile = tid >> 5;                                 // mt*32 + kt
  int kt = tile & 31, mt = tile >> 5;
  int row = mt * 16 + (lane & 15);
  const float* src = x + (size_t)row * I_SIZE + kt * 32;
  unsigned short* d = dst + (size_t)tile * 512 + lane * 16;
#pragma unroll
  for (int e = 0; e < 16; ++e) d[e] = f2bf(src[kofA(lane, e)]);
}

// Pack concat[w0|w1|w2|w3] (each 1024x1024 f32) into bf16 B-tiles (32x16).
__global__ void pack_w_kernel(const float* __restrict__ w0,
                              const float* __restrict__ w1,
                              const float* __restrict__ w2,
                              const float* __restrict__ w3,
                              unsigned short* __restrict__ dst) {
  int tid  = blockIdx.x * blockDim.x + threadIdx.x;   // 262,144 threads
  int lane = tid & 31;
  int tile = tid >> 5;                                 // nt*32 + kt
  int kt = tile & 31, nt = tile >> 5;
  int n = nt * 16 + (lane & 15);
  int gate = n >> 10, col = n & 1023;
  const float* w = (gate == 0) ? w0 : (gate == 1) ? w1 : (gate == 2) ? w2 : w3;
  int kbase = kt * 32 + ((lane >= 16) ? 8 : 0);
  unsigned short* d = dst + (size_t)tile * 512 + lane * 16;
#pragma unroll
  for (int e = 0; e < 16; ++e) d[e] = f2bf(w[(size_t)(kbase + e + ((e >= 8) ? 8 : 0)) * H_SIZE + col]);
}

// Repack current h (64x1024 f32) into bf16 A-tiles for the next step.
__global__ void pack_h_kernel(const float* __restrict__ h,
                              unsigned short* __restrict__ dst) {
  int tid  = blockIdx.x * blockDim.x + threadIdx.x;   // 4096 threads
  int lane = tid & 31;
  int tile = tid >> 5;                                 // mt*32 + kt (mt<4)
  int kt = tile & 31, mt = tile >> 5;
  int row = mt * 16 + (lane & 15);
  const float* src = h + (size_t)row * H_SIZE + kt * 32;
  unsigned short* d = dst + (size_t)tile * 512 + lane * 16;
#pragma unroll
  for (int e = 0; e < 16; ++e) d[e] = f2bf(src[kofA(lane, e)]);
}

__global__ void init_state_kernel(float* __restrict__ h, float* __restrict__ c,
                                  unsigned short* __restrict__ hp) {
  int i = blockIdx.x * blockDim.x + threadIdx.x;
  if (i < BATCH * H_SIZE) { h[i] = 0.0f; c[i] = 0.0f; hp[i] = 0; }
}

// ---------- GEMM 1: x_proj = x @ Wi + (b_i* + b_h*) ----------
__global__ void __launch_bounds__(256)
gemm1_kernel(const unsigned short* __restrict__ pX,
             const unsigned short* __restrict__ pWi,
             const float* __restrict__ b_ii, const float* __restrict__ b_hi,
             const float* __restrict__ b_if, const float* __restrict__ b_hf,
             const float* __restrict__ b_ig, const float* __restrict__ b_hg,
             const float* __restrict__ b_io, const float* __restrict__ b_ho,
             float* __restrict__ xproj) {
  __shared__ __align__(32) unsigned short sA[32 * 512];   // 32 KB: 32 A k-tiles
  int mt    = blockIdx.x >> 3;            // 0..2047
  int strip = blockIdx.x & 7;             // 512-col strip
  int wave  = threadIdx.x >> 5;
  int lane  = threadIdx.x & 31;

  {   // async global->LDS stage of the block's 32 A-tiles (ASYNCcnt path)
    const char* gsrc = (const char*)(pX + (size_t)mt * 32 * 512);
    unsigned int lbase = (unsigned int)(size_t)(&sA[0]);
#pragma unroll
    for (int i = 0; i < 8; ++i) {
      int idx = threadIdx.x + i * 256;             // 2048 x 16B = 32 KB
      async_ld_lds_b128(lbase + idx * 16, gsrc + (size_t)idx * 16);
    }
    wait_asynccnt0();
  }
  __syncthreads();

  int nt0 = strip * 32 + wave * 4;        // 4 consecutive N-tiles per wave
  const unsigned short* bbase = pWi + ((size_t)(nt0 * 32) * 32 + lane) * 16;
  v8f a0 = {}, a1 = {}, a2 = {}, a3 = {};

  // software-pipelined: prefetch kt+1 operands while WMMAs consume kt
  v16bf av = *(const v16bf*)(sA + lane * 16);
  v16bf b0 = *(const v16bf*)(bbase);
  v16bf b1 = *(const v16bf*)(bbase + 16384);
  v16bf b2 = *(const v16bf*)(bbase + 32768);
  v16bf b3 = *(const v16bf*)(bbase + 49152);
#pragma unroll 4
  for (int kt = 0; kt < 32; ++kt) {
    v16bf nav = av, nb0 = b0, nb1 = b1, nb2 = b2, nb3 = b3;
    if (kt < 31) {
      const unsigned short* bp = bbase + (kt + 1) * 512;
      nav = *(const v16bf*)(sA + ((kt + 1) * 32 + lane) * 16);
      nb0 = *(const v16bf*)(bp);
      nb1 = *(const v16bf*)(bp + 16384);
      nb2 = *(const v16bf*)(bp + 32768);
      nb3 = *(const v16bf*)(bp + 49152);
    }
    a0 = wmma_bf16(av, b0, a0);
    a1 = wmma_bf16(av, b1, a1);
    a2 = wmma_bf16(av, b2, a2);
    a3 = wmma_bf16(av, b3, a3);
    av = nav; b0 = nb0; b1 = nb1; b2 = nb2; b3 = nb3;
  }

  int l15 = lane & 15;
  int n0 = (nt0 + 0) * 16 + l15, n1 = (nt0 + 1) * 16 + l15;
  int n2 = (nt0 + 2) * 16 + l15, n3 = (nt0 + 3) * 16 + l15;
  float bs0 = bias_sum(n0, b_ii, b_hi, b_if, b_hf, b_ig, b_hg, b_io, b_ho);
  float bs1 = bias_sum(n1, b_ii, b_hi, b_if, b_hf, b_ig, b_hg, b_io, b_ho);
  float bs2 = bias_sum(n2, b_ii, b_hi, b_if, b_hf, b_ig, b_hg, b_io, b_ho);
  float bs3 = bias_sum(n3, b_ii, b_hi, b_if, b_hf, b_ig, b_hg, b_io, b_ho);
  int rowBase = mt * 16 + ((lane >= 16) ? 8 : 0);   // C-layout: VGPR r -> M=r(+8)
#pragma unroll
  for (int r = 0; r < 8; ++r) {
    size_t ro = (size_t)(rowBase + r) * N4H;
    xproj[ro + n0] = a0[r] + bs0;
    xproj[ro + n1] = a1[r] + bs1;
    xproj[ro + n2] = a2[r] + bs2;
    xproj[ro + n3] = a3[r] + bs3;
  }
}

// ---------- fused recurrent step: gates = x_proj[t] + h @ Wh ; LSTM cell ----
__global__ void __launch_bounds__(256)
lstm_step_kernel(const unsigned short* __restrict__ pWh,
                 const unsigned short* __restrict__ hpack,
                 const float* __restrict__ xproj,
                 float* __restrict__ h, float* __restrict__ c, int t) {
  __shared__ __align__(32) unsigned short sA[32 * 512];
  int W    = blockIdx.x * 8 + (threadIdx.x >> 5);
  int mt   = W >> 6;                      // constant within a block
  int ht   = W & 63;
  int lane = threadIdx.x & 31;

  {   // async global->LDS stage of packed h tiles for this batch tile
    const char* gsrc = (const char*)(hpack + (size_t)mt * 32 * 512);
    unsigned int lbase = (unsigned int)(size_t)(&sA[0]);
#pragma unroll
    for (int i = 0; i < 8; ++i) {
      int idx = threadIdx.x + i * 256;
      async_ld_lds_b128(lbase + idx * 16, gsrc + (size_t)idx * 16);
    }
    wait_asynccnt0();
  }
  __syncthreads();

  const unsigned short* bbase = pWh + ((size_t)(ht * 32) * 32 + lane) * 16;
  v8f ai = {}, af = {}, ag = {}, ao = {};

  v16bf av = *(const v16bf*)(sA + lane * 16);
  v16bf bi_ = *(const v16bf*)(bbase);
  v16bf bf_ = *(const v16bf*)(bbase + 1048576);   // gate stride = 64*32*512
  v16bf bg_ = *(const v16bf*)(bbase + 2097152);
  v16bf bo_ = *(const v16bf*)(bbase + 3145728);
#pragma unroll 4
  for (int kt = 0; kt < 32; ++kt) {
    v16bf nav = av, ni = bi_, nf = bf_, ng = bg_, no = bo_;
    if (kt < 31) {
      const unsigned short* bp = bbase + (kt + 1) * 512;
      nav = *(const v16bf*)(sA + ((kt + 1) * 32 + lane) * 16);
      ni = *(const v16bf*)(bp);
      nf = *(const v16bf*)(bp + 1048576);
      ng = *(const v16bf*)(bp + 2097152);
      no = *(const v16bf*)(bp + 3145728);
    }
    ai = wmma_bf16(av, bi_, ai);
    af = wmma_bf16(av, bf_, af);
    ag = wmma_bf16(av, bg_, ag);
    ao = wmma_bf16(av, bo_, ao);
    av = nav; bi_ = ni; bf_ = nf; bg_ = ng; bo_ = no;
  }

  int col     = ht * 16 + (lane & 15);
  int rowBase = mt * 16 + ((lane >= 16) ? 8 : 0);
#pragma unroll
  for (int r = 0; r < 8; ++r) {
    int row = rowBase + r;                               // batch index
    size_t xb = ((size_t)row * SEQ + t) * N4H;
    float gi = ai[r] + xproj[xb + col];
    float gf = af[r] + xproj[xb + 1024 + col];
    float gg = ag[r] + xproj[xb + 2048 + col];
    float go = ao[r] + xproj[xb + 3072 + col];
    float iv = sigmoid_fast(gi);
    float fv = sigmoid_fast(gf);
    float gv = tanh_fast(gg);
    float ov = sigmoid_fast(go);
    size_t ci = (size_t)row * H_SIZE + col;
    float cn = c[ci] * fv + iv * gv;
    c[ci] = cn;
    h[ci] = ov * tanh_fast(cn);
  }
}

__global__ void final_copy_kernel(const float* __restrict__ h,
                                  const float* __restrict__ c,
                                  float* __restrict__ out) {
  int i = blockIdx.x * blockDim.x + threadIdx.x;
  if (i < BATCH * H_SIZE) { out[i] = h[i]; out[BATCH * H_SIZE + i] = c[i]; }
}

// ---------- host ----------

extern "C" void kernel_launch(void* const* d_in, const int* in_sizes, int n_in,
                              void* d_out, int out_size, void* d_ws, size_t ws_size,
                              hipStream_t stream) {
  (void)in_sizes; (void)n_in; (void)out_size; (void)ws_size;
  const float* x    = (const float*)d_in[0];
  const float* w_ii = (const float*)d_in[1];
  const float* b_ii = (const float*)d_in[2];
  const float* w_hi = (const float*)d_in[3];
  const float* b_hi = (const float*)d_in[4];
  const float* w_if = (const float*)d_in[5];
  const float* b_if = (const float*)d_in[6];
  const float* w_hf = (const float*)d_in[7];
  const float* b_hf = (const float*)d_in[8];
  const float* w_ig = (const float*)d_in[9];
  const float* b_ig = (const float*)d_in[10];
  const float* w_hg = (const float*)d_in[11];
  const float* b_hg = (const float*)d_in[12];
  const float* w_io = (const float*)d_in[13];
  const float* b_io = (const float*)d_in[14];
  const float* w_ho = (const float*)d_in[15];
  const float* b_ho = (const float*)d_in[16];

  char* ws = (char*)d_ws;
  size_t o = 0;
  unsigned short* pWi = (unsigned short*)(ws + o); o += (size_t)N4H * I_SIZE * 2;   // 8 MB
  unsigned short* pWh = (unsigned short*)(ws + o); o += (size_t)N4H * H_SIZE * 2;   // 8 MB
  unsigned short* pX  = (unsigned short*)(ws + o); o += (size_t)MTOT * I_SIZE * 2;  // 64 MB
  float* xproj        = (float*)(ws + o);          o += (size_t)MTOT * N4H * 4;     // 512 MB
  float* hbuf         = (float*)(ws + o);          o += (size_t)BATCH * H_SIZE * 4;
  float* cbuf         = (float*)(ws + o);          o += (size_t)BATCH * H_SIZE * 4;
  unsigned short* hp  = (unsigned short*)(ws + o); o += (size_t)BATCH * H_SIZE * 2;

  pack_w_kernel<<<1024, 256, 0, stream>>>(w_ii, w_if, w_ig, w_io, pWi);
  pack_w_kernel<<<1024, 256, 0, stream>>>(w_hi, w_hf, w_hg, w_ho, pWh);
  pack_x_kernel<<<8192, 256, 0, stream>>>(x, pX);
  gemm1_kernel<<<16384, 256, 0, stream>>>(pX, pWi, b_ii, b_hi, b_if, b_hf,
                                          b_ig, b_hg, b_io, b_ho, xproj);
  init_state_kernel<<<256, 256, 0, stream>>>(hbuf, cbuf, hp);

  for (int t = 0; t < SEQ; ++t) {
    lstm_step_kernel<<<32, 256, 0, stream>>>(pWh, hp, xproj, hbuf, cbuf, t);
    pack_h_kernel<<<16, 256, 0, stream>>>(hbuf, hp);
  }
  final_copy_kernel<<<256, 256, 0, stream>>>(hbuf, cbuf, (float*)d_out);
}
